// CausalSelfAttention_6373731467737
// MI455X (gfx1250) — matrix operand order repaired and checked by reading, not compiled, not run on hardware
//
#include <hip/hip_runtime.h>
#include <hip/hip_bf16.h>
#include <math.h>
#include <stdint.h>

typedef _Float16 h16;
typedef __attribute__((ext_vector_type(16))) _Float16 v16h;
typedef __attribute__((ext_vector_type(8)))  _Float16 v8h;
typedef __attribute__((ext_vector_type(8)))  float    v8f;
typedef __attribute__((ext_vector_type(4)))  float    v4f;
typedef __attribute__((ext_vector_type(4)))  unsigned int u32x4;
typedef __attribute__((ext_vector_type(8)))  int      i32x8;
typedef __attribute__((ext_vector_type(4)))  int      i32x4;

#define BB   2
#define TT   2048
#define EMB  4096
#define NHQ  32
#define NHKV 8
#define HDIM 128
#define BT   (BB*TT)

union FragA { v16h v; v8h h[2]; };

__device__ __forceinline__ v8f wmma16(v16h a, v16h b, v8f c) {
  // 8 args: (neg_a, A, neg_b, B, c_mod, C, reuse_a, reuse_b)
  return __builtin_amdgcn_wmma_f32_16x16x32_f16(false, a, false, b, (short)0, c,
                                                false, false);
}

// ---------------------------------------------------------------------------
// Tensor Data Mover: DMA a 2-D f16 tile (tileRows x 32 elems, row stride
// `strideElems`) from global into LDS at byte offset `ldsOff`.
// D# layout per CDNA5 ISA 8.3/8.4: group0 = {count/flags, lds_addr,
// global_addr[31:0], global_addr[56:32]|type=2}; group1 packs data_size,
// tensor dims, tile dims, dim0 stride. This toolchain's builtin takes the
// 6-arg form: (g0, g1, g2, g3, extra, cpol).
// ---------------------------------------------------------------------------
__device__ __forceinline__ void tdm_load_tile_f16(const h16* gptr,
                                                  uint32_t ldsOff,
                                                  uint32_t strideElems,
                                                  uint32_t tensorRows,
                                                  uint32_t tileRows) {
  const uint64_t ga = (uint64_t)(uintptr_t)gptr;
  u32x4 g0;
  g0[0] = 1u;                                            // count=1, user mode
  g0[1] = ldsOff;                                        // lds_addr (bytes)
  g0[2] = (uint32_t)ga;                                  // global_addr[31:0]
  g0[3] = (uint32_t)((ga >> 32) & 0x01FFFFFFu) | (2u << 30); // [56:32] | type=2
  i32x8 g1;
  g1[0] = (int)(1u << 16);                               // data_size=1 (2B)
  g1[1] = (int)((strideElems & 0xFFFFu) << 16);          // tensor_dim0 lo16
  g1[2] = (int)((strideElems >> 16) & 0xFFFFu)           // tensor_dim0 hi16
        | (int)((tensorRows & 0xFFFFu) << 16);           // tensor_dim1 lo16
  g1[3] = (int)((tensorRows >> 16) & 0xFFFFu)            // tensor_dim1 hi16
        | (int)(32u << 16);                              // tile_dim0 = 32
  g1[4] = (int)(tileRows & 0xFFFFu);                     // tile_dim1, tile_dim2=0
  g1[5] = (int)strideElems;                              // dim0_stride lo32
  g1[6] = 0;                                             // dim0_stride hi16, d1s lo
  g1[7] = 0;
  i32x4 gz4 = {0, 0, 0, 0};
  i32x8 gz8 = {0, 0, 0, 0, 0, 0, 0, 0};
  __builtin_amdgcn_tensor_load_to_lds(g0, g1, gz4, gz4, gz8, 0);
}

// ---------------------------------------------------------------------------
// Tiled WMMA GEMM: C[M x N] = A[M x K] * Bw[K x N]
//   A_F32 : A is fp32 (converted to f16 while staging) else f16
//   TDM_A : stage the (f16) A tile with the Tensor Data Mover
//   OMODE : 0 = f16 row-major (Q/K) | 1 = f16 [b][kvh][hd][t] (V^T)
//           2 = f32 row-major (final output)
// Block 256 threads / 8 waves, tile 128x128, k-step 32, double-buffered LDS.
// ---------------------------------------------------------------------------
template<bool A_F32, int OMODE, bool TDM_A>
__global__ __launch_bounds__(256) void gemm_wmma_kernel(
    const void* __restrict__ Ap, const float* __restrict__ Bw,
    void* __restrict__ Cp, int M, int N, int K, int ldc)
{
  __shared__ __align__(16) h16 As[2][128*32];   // [m][k]
  __shared__ __align__(16) h16 Bs[2][128*32];   // transposed [n][k]

  const int tid = threadIdx.x;
  const int bx = blockIdx.x, by = blockIdx.y;
  const int wid = tid >> 5, lane = tid & 31;
  const int lr = lane & 15, hi = lane >> 4;
  const int wm = wid >> 2;            // 0..1 : 64-row half
  const int wn = wid & 3;             // 0..3 : 32-col quarter

  const int am = tid >> 1,  ak = (tid & 1) * 16;   // A staging coords
  const int bk = tid >> 3,  bn = (tid & 7) * 16;   // B staging coords

  v8f acc[4][2];
  #pragma unroll
  for (int i = 0; i < 4; i++) { acc[i][0] = {}; acc[i][1] = {}; }

  const float* aRowF = (const float*)Ap + (size_t)(by*128 + am) * K;
  const h16*   aRowH = (const h16*)Ap   + (size_t)(by*128 + am) * K;
  const h16*   aTdmBase = (const h16*)Ap + (size_t)(by*128) * K;

  float aR[16];         // f32-A staging registers
  v8h   aH[2];          // f16-A staging registers
  float bR[16];         // B staging registers

  // ---------------- prologue: stage k0 = 0 into buffer 0 ------------------
  if constexpr (TDM_A) {
    if (wid == 0)
      tdm_load_tile_f16(aTdmBase, (uint32_t)(uintptr_t)&As[0][0],
                        (uint32_t)K, (uint32_t)M, 128u);
  } else if constexpr (A_F32) {
    #pragma unroll
    for (int i = 0; i < 4; i++) {
      v4f f = *(const v4f*)(aRowF + ak + i*4);
      #pragma unroll
      for (int j = 0; j < 4; j++) As[0][am*32 + ak + i*4 + j] = (h16)f[j];
    }
  } else {
    *(v8h*)&As[0][am*32 + ak]     = *(const v8h*)(aRowH + ak);
    *(v8h*)&As[0][am*32 + ak + 8] = *(const v8h*)(aRowH + ak + 8);
  }
  {
    const float* brow = Bw + (size_t)bk * N + bx*128 + bn;
    #pragma unroll
    for (int i = 0; i < 4; i++) {
      v4f f = *(const v4f*)(brow + i*4);
      #pragma unroll
      for (int j = 0; j < 4; j++) Bs[0][(bn + i*4 + j)*32 + bk] = (h16)f[j];
    }
  }
  if constexpr (TDM_A) { if (wid == 0) __builtin_amdgcn_s_wait_tensorcnt(0); }
  __syncthreads();

  // ---------------- main loop (double-buffered) ---------------------------
  int buf = 0;
  for (int k0 = 0; k0 < K; k0 += 32) {
    const bool haveNext = (k0 + 32 < K);

    // issue next tile's loads (overlap with this tile's WMMAs)
    if (haveNext) {
      const int kn = k0 + 32;
      if constexpr (TDM_A) {
        if (wid == 0)
          tdm_load_tile_f16(aTdmBase + kn,
                            (uint32_t)(uintptr_t)&As[buf ^ 1][0],
                            (uint32_t)K, (uint32_t)M, 128u);
      } else if constexpr (A_F32) {
        #pragma unroll
        for (int i = 0; i < 4; i++) {
          v4f f = *(const v4f*)(aRowF + kn + ak + i*4);
          #pragma unroll
          for (int j = 0; j < 4; j++) aR[i*4 + j] = f[j];
        }
      } else {
        aH[0] = *(const v8h*)(aRowH + kn + ak);
        aH[1] = *(const v8h*)(aRowH + kn + ak + 8);
      }
      {
        const float* brow = Bw + (size_t)(kn + bk) * N + bx*128 + bn;
        #pragma unroll
        for (int i = 0; i < 4; i++) {
          v4f f = *(const v4f*)(brow + i*4);
          #pragma unroll
          for (int j = 0; j < 4; j++) bR[i*4 + j] = f[j];
        }
      }
      if (k0 + 64 < K) {   // prefetch two steps ahead -> global_prefetch_b8
        if constexpr (A_F32)
          __builtin_prefetch(aRowF + k0 + 64 + ak, 0, 1);
        else if constexpr (!TDM_A)
          __builtin_prefetch(aRowH + k0 + 64 + ak, 0, 1);
        __builtin_prefetch(Bw + (size_t)(k0 + 64 + bk) * N + bx*128 + bn, 0, 1);
      }
    }

    // compute from current buffer
    FragA afr[4];
    #pragma unroll
    for (int mt = 0; mt < 4; mt++) {
      const int row = wm*64 + mt*16 + lr;
      afr[mt].h[0] = *(const v8h*)&As[buf][row*32 + hi*8];
      afr[mt].h[1] = *(const v8h*)&As[buf][row*32 + 16 + hi*8];
    }
    #pragma unroll
    for (int nt = 0; nt < 2; nt++) {
      const int col = wn*32 + nt*16 + lr;
      FragA bfr;
      bfr.h[0] = *(const v8h*)&Bs[buf][col*32 + hi*8];
      bfr.h[1] = *(const v8h*)&Bs[buf][col*32 + 16 + hi*8];
      #pragma unroll
      for (int mt = 0; mt < 4; mt++)
        acc[mt][nt] = wmma16(afr[mt].v, bfr.v, acc[mt][nt]);
    }

    // commit staged registers into the other buffer
    if (haveNext) {
      if constexpr (!TDM_A) {
        if constexpr (A_F32) {
          #pragma unroll
          for (int i = 0; i < 16; i++) As[buf^1][am*32 + ak + i] = (h16)aR[i];
        } else {
          *(v8h*)&As[buf^1][am*32 + ak]     = aH[0];
          *(v8h*)&As[buf^1][am*32 + ak + 8] = aH[1];
        }
      }
      #pragma unroll
      for (int i = 0; i < 16; i++) Bs[buf^1][(bn + i)*32 + bk] = (h16)bR[i];
      if constexpr (TDM_A) {
        if (wid == 0) __builtin_amdgcn_s_wait_tensorcnt(0);
      }
    }
    __syncthreads();
    buf ^= 1;
  }

  // ---- epilogue: C/D layout lane=n, vgpr r -> m = r + hi*8 ----
  #pragma unroll
  for (int mt = 0; mt < 4; mt++) {
    #pragma unroll
    for (int nt = 0; nt < 2; nt++) {
      const int mbase = by*128 + wm*64 + mt*16 + hi*8;
      const int ncol  = bx*128 + wn*32 + nt*16 + lr;
      if constexpr (OMODE == 0) {
        h16* C = (h16*)Cp;
        #pragma unroll
        for (int r = 0; r < 8; r++)
          C[(size_t)(mbase + r)*ldc + ncol] = (h16)acc[mt][nt][r];
      } else if constexpr (OMODE == 1) {
        // V^T: [b][kvh][hd][t]; 8 accum rows = 8 consecutive t -> one b128
        h16* C = (h16*)Cp;
        const int kvh = ncol >> 7, hd = ncol & 127;
        const int bb = mbase >> 11, t0 = mbase & (TT - 1);
        v8h p;
        #pragma unroll
        for (int r = 0; r < 8; r++) p[r] = (h16)acc[mt][nt][r];
        *(v8h*)&C[(((size_t)bb*NHKV + kvh)*HDIM + hd)*TT + t0] = p;
      } else {
        float* C = (float*)Cp;
        #pragma unroll
        for (int r = 0; r < 8; r++)
          C[(size_t)(mbase + r)*ldc + ncol] = acc[mt][nt][r];
      }
    }
  }
}

// ---------------------------------------------------------------------------
// RoPE (half-split rotate), in place on f16 row-major [B*T][nheads*128].
// ---------------------------------------------------------------------------
__global__ __launch_bounds__(256) void rope_kernel(h16* __restrict__ X,
                                                   int nheads, int total)
{
  const int idx = blockIdx.x * 256 + threadIdx.x;
  if (idx >= total) return;
  const int d   = idx & 63;
  const int h   = (idx >> 6) % nheads;
  const int row = idx / (64 * nheads);
  const int t   = row & (TT - 1);
  const size_t base = (size_t)row * (nheads * HDIM) + h * HDIM + d;
  const float x1 = (float)X[base];
  const float x2 = (float)X[base + 64];
  const float ang = (float)t * __expf(-0.14391156f * (float)d); // 10000^(-d/64)
  float s, c;
  __sincosf(ang, &s, &c);
  X[base]      = (h16)(x1 * c - x2 * s);
  X[base + 64] = (h16)(x2 * c + x1 * s);
}

// ---------------------------------------------------------------------------
// Flash attention (causal, GQA 4:1). Block = 128 threads = 4 waves.
// Wave w owns query rows qt*64 + w*16 .. +15; iterates 32-key blocks.
// K/V fragments are batch-loaded into distinct registers before the WMMA
// chain so one loadcnt wait covers a whole clause of b128 loads.
// ---------------------------------------------------------------------------
__global__ __launch_bounds__(128) void attn_kernel(
    const h16* __restrict__ Q, const h16* __restrict__ Kh,
    const h16* __restrict__ Vt, h16* __restrict__ O)
{
  __shared__ __align__(16) float Ss[4][16][32];
  __shared__ __align__(16) h16   Ps[4][16][32];
  __shared__ float Cs[4][16];
  __shared__ float Ls[4][16];

  const int qt = blockIdx.x, h = blockIdx.y, b = blockIdx.z;
  const int wid = threadIdx.x >> 5, lane = threadIdx.x & 31;
  const int lr = lane & 15, hi = lane >> 4;
  const int kvh = h >> 2;                 // HQ/HKV = 4
  const int q0 = qt*64 + wid*16;
  const float scale = 0.08838834764831845f;   // 1/sqrt(128)

  // Q fragments: 16 rows x 128 (4 k-chunks of 32), kept in registers
  FragA qf[4];
  {
    const h16* qrow = Q + (size_t)(b*TT + q0 + lr)*EMB + h*HDIM;
    #pragma unroll
    for (int j = 0; j < 4; j++) {
      qf[j].h[0] = *(const v8h*)(qrow + j*32 + hi*8);
      qf[j].h[1] = *(const v8h*)(qrow + j*32 + 16 + hi*8);
    }
  }

  v8f o[8];
  #pragma unroll
  for (int j = 0; j < 8; j++) o[j] = {};
  float m_r = -3.0e38f, l_r = 0.0f;       // per-row stats (lanes 0..15)

  const int nkb = qt*2 + 2;               // key blocks covering keys <= qmax
  for (int kb = 0; kb < nkb; kb++) {
    // ---- S = Q K^T over 32 keys (2 n-tiles x 4 k-steps) ----
    v8f s[2]; s[0] = {}; s[1] = {};
    #pragma unroll
    for (int nt = 0; nt < 2; nt++) {
      const int key = kb*32 + nt*16 + lr;
      const h16* kp = Kh + (size_t)(b*TT + key)*(NHKV*HDIM) + kvh*HDIM;
      FragA bf[4];                        // batch all 4 k-step fragments
      #pragma unroll
      for (int ks = 0; ks < 4; ks++) {
        bf[ks].h[0] = *(const v8h*)(kp + ks*32 + hi*8);
        bf[ks].h[1] = *(const v8h*)(kp + ks*32 + 16 + hi*8);
      }
      if (kb + 1 < nkb)                   // prefetch next key block's row
        __builtin_prefetch(kp + (size_t)32*(NHKV*HDIM), 0, 1);
      #pragma unroll
      for (int ks = 0; ks < 4; ks++)
        s[nt] = wmma16(qf[ks].v, bf[ks].v, s[nt]);
    }
    // ---- spill S to LDS (lane=n, vgpr r -> m = r + hi*8) ----
    #pragma unroll
    for (int nt = 0; nt < 2; nt++)
      #pragma unroll
      for (int r = 0; r < 8; r++)
        Ss[wid][hi*8 + r][nt*16 + lr] = s[nt][r];
    __syncthreads();

    // ---- online softmax: lanes 0..15 each own one query row ----
    if (lane < 16) {
      const int qrow  = q0 + lane;
      const int valid = qrow - kb*32 + 1;        // unmasked keys in block
      const int vn    = valid < 32 ? valid : 32;
      float mx = m_r;
      for (int j = 0; j < vn; j++)
        mx = fmaxf(mx, Ss[wid][lane][j] * scale);
      const float c = __expf(m_r - mx);
      float sum = 0.0f;
      for (int j = 0; j < 32; j++) {
        const float p = (j < vn) ? __expf(Ss[wid][lane][j]*scale - mx) : 0.0f;
        sum += p;
        Ps[wid][lane][j] = (h16)p;
      }
      l_r = l_r * c + sum;
      m_r = mx;
      Cs[wid][lane] = c;
    }
    __syncthreads();

    // ---- rescale O, then O += P * V ----
    float cr[8];
    #pragma unroll
    for (int r = 0; r < 8; r++) cr[r] = Cs[wid][hi*8 + r];
    #pragma unroll
    for (int jt = 0; jt < 8; jt++)
      #pragma unroll
      for (int r = 0; r < 8; r++) o[jt][r] *= cr[r];

    FragA pf;                                   // P as A-frag (16 x 32 keys)
    pf.h[0] = *(const v8h*)&Ps[wid][lr][hi*8];
    pf.h[1] = *(const v8h*)&Ps[wid][lr][16 + hi*8];

    const h16* vbase = Vt + ((size_t)b*NHKV + kvh)*HDIM*TT + kb*32 + hi*8;
    #pragma unroll
    for (int g = 0; g < 2; g++) {               // 2 groups of 4 hd-tiles
      FragA vf[4];
      #pragma unroll
      for (int j = 0; j < 4; j++) {
        const h16* vrow = vbase + (size_t)((g*4 + j)*16 + lr)*TT;
        vf[j].h[0] = *(const v8h*)(vrow);
        vf[j].h[1] = *(const v8h*)(vrow + 16);
      }
      #pragma unroll
      for (int j = 0; j < 4; j++)
        o[g*4 + j] = wmma16(pf.v, vf[j].v, o[g*4 + j]);
    }
    __syncthreads();
  }

  // ---- normalize and store [B*T][4096] f16 ----
  if (lane < 16) Ls[wid][lane] = l_r;
  __syncthreads();
  float rl[8];
  #pragma unroll
  for (int r = 0; r < 8; r++) rl[r] = 1.0f / Ls[wid][hi*8 + r];
  #pragma unroll
  for (int jt = 0; jt < 8; jt++) {
    #pragma unroll
    for (int r = 0; r < 8; r++) {
      const int trow = q0 + hi*8 + r;
      O[(size_t)(b*TT + trow)*EMB + h*HDIM + jt*16 + lr] = (h16)(o[jt][r]*rl[r]);
    }
  }
}

// ---------------------------------------------------------------------------
extern "C" void kernel_launch(void* const* d_in, const int* in_sizes, int n_in,
                              void* d_out, int out_size, void* d_ws, size_t ws_size,
                              hipStream_t stream) {
  const float* x  = (const float*)d_in[0];
  const float* wq = (const float*)d_in[1];
  const float* wk = (const float*)d_in[2];
  const float* wv = (const float*)d_in[3];
  const float* wp = (const float*)d_in[4];
  float* out = (float*)d_out;

  // f16 workspace: Q (32MB) | K (8MB) | V^T (8MB) | attn-out (32MB) = 80MB
  h16* Qh = (h16*)d_ws;
  h16* Kh = Qh + (size_t)BT * EMB;
  h16* Vt = Kh + (size_t)BT * (NHKV*HDIM);
  h16* Ah = Vt + (size_t)BT * (NHKV*HDIM);

  // Q = x @ wq  -> f16 [B*T][4096]
  gemm_wmma_kernel<true, 0, false><<<dim3(EMB/128, BT/128), 256, 0, stream>>>(
      x, wq, Qh, BT, EMB, EMB, EMB);
  // K = x @ wk  -> f16 [B*T][1024]
  gemm_wmma_kernel<true, 0, false><<<dim3((NHKV*HDIM)/128, BT/128), 256, 0, stream>>>(
      x, wk, Kh, BT, NHKV*HDIM, EMB, NHKV*HDIM);
  // V = x @ wv  -> f16 transposed [b][kvh][hd][t]
  gemm_wmma_kernel<true, 1, false><<<dim3((NHKV*HDIM)/128, BT/128), 256, 0, stream>>>(
      x, wv, Vt, BT, NHKV*HDIM, EMB, 0);

  { const int total = BT * NHQ  * 64;
    rope_kernel<<<(total + 255)/256, 256, 0, stream>>>(Qh, NHQ,  total); }
  { const int total = BT * NHKV * 64;
    rope_kernel<<<(total + 255)/256, 256, 0, stream>>>(Kh, NHKV, total); }

  attn_kernel<<<dim3(TT/64, NHQ, BB), 128, 0, stream>>>(Qh, Kh, Vt, Ah);

  // out = attn @ wproj -> f32 [B*T][4096]; A tile staged via Tensor Data Mover
  gemm_wmma_kernel<false, 2, true><<<dim3(EMB/128, BT/128), 256, 0, stream>>>(
      Ah, wp, out, BT, EMB, EMB, EMB);
}
